// InventoryEncoder_90065464197721
// MI455X (gfx1250) — matrix operand order repaired and checked by reading, not compiled, not run on hardware
//
#include <hip/hip_runtime.h>
#include <math.h>

// ---------------- problem constants (from reference) ----------------
#define B_N    1024
#define INV    55
#define SLEN   80
#define EMB_D  16
#define HID    16
#define OUT_D  24
#define EOS_TK 129
#define NSTR   (B_N * INV)        // 56320 strings
#define NGROUP (NSTR / 16)        // 3520 wave-groups of 16 strings
#define GROUPS_PER_BLK 4          // 8 waves/block = 4 groups x 2 directions

// d_out layout (floats): [inv_features | inv_class_emb | str_emb]
#define OUT2_OFF (B_N * OUT_D)                    // 24576
#define OUT3_OFF (OUT2_OFF + B_N * INV * EMB_D)   // 925696

// d_ws layout (floats): [h_class (1024*16) | h_str (56320*32)]
#define WS_HSTR_OFF (B_N * HID)                   // 16384

typedef __attribute__((ext_vector_type(16))) _Float16 v16h;
typedef __attribute__((ext_vector_type(8)))  _Float16 v8h;
typedef __attribute__((ext_vector_type(8)))  float    v8f;

__device__ __constant__ int kMap[19] =
    {128, 93, 41, 91, 61, 34, 40, 37, 33, 63, 43, 47, 36, 42, 96, 48, 95, 46, 128};

// ---------------- fast hardware transcendentals ----------------
// gfx1250 has V_TANH_F32 (ISA TRANS op list). clang-22 lacks the builtin,
// so fall back to inline asm; trailing v_nop satisfies the TRANS-op
// result/source hazard rule (compiler cannot classify inline asm as TRANS).
__device__ __forceinline__ float fast_tanh(float x) {
#if __has_builtin(__builtin_amdgcn_tanhf)
    return __builtin_amdgcn_tanhf(x);
#elif __has_builtin(__builtin_amdgcn_tanh_f32)
    return __builtin_amdgcn_tanh_f32(x);
#else
    float r;
    asm volatile("v_tanh_f32 %0, %1\n\tv_nop" : "=v"(r) : "v"(x));
    return r;
#endif
}
// sigmoid(x) = 0.5*tanh(0.5x) + 0.5  ->  fma + v_tanh + fma
__device__ __forceinline__ float fast_sigmoid(float x) {
    return __builtin_fmaf(0.5f, fast_tanh(0.5f * x), 0.5f);
}

// =====================================================================
// Kernel A: object-class branch. 16 batch rows / block of 256 threads,
// thread (row, j) owns hidden unit j. Small (58 MFLOP) -> VALU.
// =====================================================================
__global__ void __launch_bounds__(256) class_rnn_kernel(
    const int* __restrict__ ocl, const float* __restrict__ emb,
    const float* __restrict__ Wih, const float* __restrict__ Whh,
    const float* __restrict__ bih, const float* __restrict__ bhh,
    float* __restrict__ out2, float* __restrict__ hcls)
{
    __shared__ float hbuf[16][16];
    const int row = threadIdx.x >> 4;
    const int j   = threadIdx.x & 15;
    const int b   = blockIdx.x * 16 + row;

    float wih[16], whh[16];
#pragma unroll
    for (int k = 0; k < 16; k++) { wih[k] = Wih[j * 16 + k]; whh[k] = Whh[j * 16 + k]; }
    const float bi = bih[j], bh = bhh[j];

    int len = 0;
    for (int i = 0; i < INV; i++) len += (ocl[b * INV + i] != 18);

    float h = 0.0f;
    hbuf[row][j] = 0.0f;
    __syncthreads();

    for (int t = 0; t < INV; t++) {
        const int cls = kMap[ocl[b * INV + t]];
        const float* e = emb + cls * 16;                 // cls in {33..128}, never padding 0
        out2[(size_t)(b * INV + t) * 16 + j] = e[j];

        float gi = bi, gh = bh;
#pragma unroll
        for (int k = 0; k < 16; k++) {
            gi += e[k] * wih[k];
            gh += hbuf[row][k] * whh[k];
        }
        const float hn = fast_tanh(gi + gh);
        if (t < len) h = hn;
        __syncthreads();
        hbuf[row][j] = h;
        __syncthreads();
    }
    hcls[b * 16 + j] = h;
}

// =====================================================================
// Kernel B: string branch. One wave32 = 16 strings x 1 direction.
// GRU step = 6x V_WMMA_F32_16X16X32_F16 (K=16 zero-padded to 32):
//   gr = X*WihT_r ; gr = H*WhhT_r + gr   (gi fused into gh via C operand)
//   gz = X*WihT_z ; gz = H*WhhT_z + gz
//   gn_i = X*WihT_n ; gn_h = H*WhhT_n    (kept split: n = tanh(i_n + r*h_n))
// biases applied as per-lane scalars in the activation inputs.
// h kept in C/D layout (8 f32 / lane); re-shaped to A layout each step
// through a 512B LDS stage (ds_store_b16 x8 + ds_load_b128; LDS ops are
// in-order within a wave, no barrier needed).
// =====================================================================
__global__ void __launch_bounds__(256, 1) str_gru_kernel(
    const int* __restrict__ strs, const float* __restrict__ emb,
    const float* __restrict__ WihF, const float* __restrict__ WhhF,
    const float* __restrict__ bihF, const float* __restrict__ bhhF,
    const float* __restrict__ WihB, const float* __restrict__ WhhB,
    const float* __restrict__ bihB, const float* __restrict__ bhhB,
    float* __restrict__ out3, float* __restrict__ hstr)
{
    __shared__ int ldsT[GROUPS_PER_BLK * 16 * SLEN];        // 20480 B tokens
    __shared__ __align__(16) _Float16 stage[8 * 256];       // 4096 B h-stage, 1/wave
    __shared__ int lenW[8 * 16];

    const int tid    = threadIdx.x;
    const int waveId = tid >> 5;
    const int lane   = tid & 31;

    // tokens of the block's 4 groups are one contiguous 20KB slab: copy coalesced
    {
        const size_t base = (size_t)blockIdx.x * GROUPS_PER_BLK * 16 * SLEN;
        for (int i = tid; i < GROUPS_PER_BLK * 16 * SLEN; i += 256)
            ldsT[i] = strs[base + i];
    }
    __syncthreads();

    const int pair  = waveId >> 1;
    const int dir   = waveId & 1;                     // 0 = forward, 1 = backward
    const int group = blockIdx.x * GROUPS_PER_BLK + pair;
    const int* toks = ldsT + pair * (16 * SLEN);
    _Float16*  stg  = stage + waveId * 256;           // [16][16] f16 row-major
    int*       lenw = lenW + waveId * 16;

    // per-string length = count of nonzero tokens
    if (lane < 16) {
        int c = 0;
        for (int t = 0; t < SLEN; t++) c += (toks[lane * SLEN + t] != 0);
        lenw[lane] = c;
    }
    __asm__ volatile("" ::: "memory");   // wave-internal LDS ordering only

    const int mm = lane & 15;            // A-layout row (string) for this lane
    const int hi = lane >> 4;            // lane half -> K half / M half
    const int kb = hi * 8;               // this lane's K-base in A layout
    const int lenm = lenw[mm];

    const float* Wih = dir ? WihB : WihF;
    const float* Whh = dir ? WhhB : WhhF;
    const float* bih = dir ? bihB : bihF;
    const float* bhh = dir ? bhhB : bhhF;

    // B-layout weights (32x16 f16, K padded): lane holds column n = mm,
    // elements 0..7 = W[c*16+n][kb..kb+7]; elements 8..15 carry the K-pad zeros.
    v16h Bi[3], Bh[3];
#pragma unroll
    for (int c = 0; c < 3; c++) {
        v16h wi = {}, wh = {};
        {
            const float* pi = Wih + (c * 16 + mm) * 16 + kb;
            const float* ph = Whh + (c * 16 + mm) * 16 + kb;
#pragma unroll
            for (int k = 0; k < 8; k++) {
                wi[k] = (_Float16)pi[k];
                wh[k] = (_Float16)ph[k];
            }
        }
        Bi[c] = wi; Bh[c] = wh;
    }

    // biases as per-lane scalars (column n = lane&15 in C/D layout):
    //   r,z: combined bih+bhh on the pre-activation; n: bih_n on input path,
    //   bhh_n inside the r*(...) term (PyTorch GRU semantics).
    const float bias_r  = bih[0 * 16 + mm] + bhh[0 * 16 + mm];
    const float bias_z  = bih[1 * 16 + mm] + bhh[1 * 16 + mm];
    const float bias_in = bih[2 * 16 + mm];
    const float bias_hn = bhh[2 * 16 + mm];

    // valid-mask lengths in D layout: VGPR v holds rows m = v + 8*hi
    int lenv[8];
#pragma unroll
    for (int v = 0; v < 8; v++) lenv[v] = lenw[v + 8 * hi];

    v8f h = {};            // hidden state, C/D layout (f32)
    const v8f czero = {};  // WMMA C operand = 0

    for (int t = 0; t < SLEN; ++t) {
        // ---------- X (A-layout): lane holds x[mm][kb..kb+7] ----------
        int t_eff = t;
        if (dir) { t_eff = lenm - 1 - t; if (t_eff < 0) t_eff = 0; }
        int tok = toks[mm * SLEN + t_eff];
        if (t_eff == lenm) tok = EOS_TK;              // EOS insertion
        float xv[8];
        if (tok == 0) {                               // padding_idx=0 -> zero row
#pragma unroll
            for (int i = 0; i < 8; i++) xv[i] = 0.0f;
        } else {
            const float* ep = emb + tok * 16 + kb;
#pragma unroll
            for (int i = 0; i < 8; i++) xv[i] = ep[i];
        }
        v16h Xa = {};
#pragma unroll
        for (int i = 0; i < 8; i++) Xa[i] = (_Float16)xv[i];
        if (dir == 0) {                               // forward wave streams str_emb out
            const size_t s = (size_t)group * 16 + mm;
            float* o = out3 + (s * SLEN + t) * 16 + kb;
#pragma unroll
            for (int i = 0; i < 8; i++) o[i] = xv[i];
        }

        // ---------- H: D layout -> A layout via LDS stage ----------
#pragma unroll
        for (int v = 0; v < 8; v++)
            stg[(v + 8 * hi) * 16 + mm] = (_Float16)h[v];
        __asm__ volatile("" ::: "memory");
        const v8h hlo = *(const v8h*)(stg + mm * 16 + hi * 8);  // ds_load_b128
        v16h Ha = {};
#pragma unroll
        for (int i = 0; i < 8; i++) Ha[i] = hlo[i];

        // ---------- 6x WMMA f32 16x16x32 f16 ----------
        v8f gr = __builtin_amdgcn_wmma_f32_16x16x32_f16(
            false, Xa, false, Bi[0], (short)0, czero, false, false);
        v8f gz = __builtin_amdgcn_wmma_f32_16x16x32_f16(
            false, Xa, false, Bi[1], (short)0, czero, false, false);
        v8f gn_i = __builtin_amdgcn_wmma_f32_16x16x32_f16(
            false, Xa, false, Bi[2], (short)0, czero, false, false);
        gr = __builtin_amdgcn_wmma_f32_16x16x32_f16(      // C-fused: gr += H*Whh_r
            false, Ha, false, Bh[0], (short)0, gr, false, false);
        gz = __builtin_amdgcn_wmma_f32_16x16x32_f16(      // C-fused: gz += H*Whh_z
            false, Ha, false, Bh[1], (short)0, gz, false, false);
        v8f gn_h = __builtin_amdgcn_wmma_f32_16x16x32_f16(
            false, Ha, false, Bh[2], (short)0, czero, false, false);

        // ---------- GRU cell elementwise in D layout (v_tanh based) ----------
#pragma unroll
        for (int v = 0; v < 8; v++) {
            const float r  = fast_sigmoid(gr[v] + bias_r);
            const float z  = fast_sigmoid(gz[v] + bias_z);
            const float n  = fast_tanh(gn_i[v] + bias_in + r * (gn_h[v] + bias_hn));
            const float hn = n + z * (h[v] - n);      // (1-z)*n + z*h
            h[v] = (t < lenv[v]) ? hn : h[v];
        }
    }

    // final hidden: h_str[s][dir*16 + n], D-layout row m = v + 8*hi, col n = mm
#pragma unroll
    for (int v = 0; v < 8; v++) {
        const size_t s = (size_t)group * 16 + (v + 8 * hi);
        hstr[s * 32 + dir * 16 + mm] = h[v];
    }
}

// =====================================================================
// Kernel C: masked mean-pool over 55 strings + 2-layer fusion MLP.
// One block of 64 threads per batch element.
// =====================================================================
__global__ void __launch_bounds__(64) fuse_kernel(
    const int* __restrict__ strs,
    const float* __restrict__ hcls, const float* __restrict__ hstr,
    const float* __restrict__ W1, const float* __restrict__ b1,
    const float* __restrict__ W2, const float* __restrict__ b2,
    float* __restrict__ out1)
{
    __shared__ float mask[INV];
    __shared__ float comb[48];
    __shared__ float c1[16];
    __shared__ float cnt;
    const int b = blockIdx.x, tid = threadIdx.x;

    if (tid < INV) {
        long s = 0;
        const int* p = strs + ((size_t)b * INV + tid) * SLEN;
        for (int t = 0; t < SLEN; t++) s += p[t];
        mask[tid] = (s > 0) ? 1.0f : 0.0f;
    }
    __syncthreads();
    if (tid == 0) { float c = 0; for (int i = 0; i < INV; i++) c += mask[i]; cnt = c; }
    if (tid < 16) comb[tid] = hcls[b * 16 + tid];
    __syncthreads();
    if (tid < 32) {
        float p = 0.0f;
        for (int i = 0; i < INV; i++)
            p += mask[i] * hstr[((size_t)b * INV + i) * 32 + tid];
        comb[16 + tid] = p / (cnt + 1e-8f);
    }
    __syncthreads();
    if (tid < 16) {
        float a = b1[tid];
#pragma unroll
        for (int k = 0; k < 48; k++) a += comb[k] * W1[tid * 48 + k];
        c1[tid] = fmaxf(a, 0.0f);
    }
    __syncthreads();
    if (tid < OUT_D) {
        float a = b2[tid];
#pragma unroll
        for (int k = 0; k < 16; k++) a += c1[k] * W2[tid * 16 + k];
        out1[b * OUT_D + tid] = a;
    }
}

// =====================================================================
extern "C" void kernel_launch(void* const* d_in, const int* in_sizes, int n_in,
                              void* d_out, int out_size, void* d_ws, size_t ws_size,
                              hipStream_t stream)
{
    const int*   ocl   = (const int*)d_in[0];
    const int*   strs  = (const int*)d_in[1];
    const float* emb   = (const float*)d_in[2];
    const float* rWih  = (const float*)d_in[3];
    const float* rWhh  = (const float*)d_in[4];
    const float* rbih  = (const float*)d_in[5];
    const float* rbhh  = (const float*)d_in[6];
    const float* gWihF = (const float*)d_in[7];
    const float* gWhhF = (const float*)d_in[8];
    const float* gbihF = (const float*)d_in[9];
    const float* gbhhF = (const float*)d_in[10];
    const float* gWihB = (const float*)d_in[11];
    const float* gWhhB = (const float*)d_in[12];
    const float* gbihB = (const float*)d_in[13];
    const float* gbhhB = (const float*)d_in[14];
    const float* fW1   = (const float*)d_in[15];
    const float* fb1   = (const float*)d_in[16];
    const float* fW2   = (const float*)d_in[17];
    const float* fb2   = (const float*)d_in[18];

    float* out  = (float*)d_out;
    float* ws   = (float*)d_ws;
    float* hcls = ws;                   // 1024*16 floats
    float* hstr = ws + WS_HSTR_OFF;     // 56320*32 floats (~7.3 MB total ws)

    class_rnn_kernel<<<B_N / 16, 256, 0, stream>>>(
        ocl, emb, rWih, rWhh, rbih, rbhh, out + OUT2_OFF, hcls);

    str_gru_kernel<<<NGROUP / GROUPS_PER_BLK, 256, 0, stream>>>(
        strs, emb, gWihF, gWhhF, gbihF, gbhhF,
        gWihB, gWhhB, gbihB, gbhhB, out + OUT3_OFF, hstr);

    fuse_kernel<<<B_N, 64, 0, stream>>>(
        strs, hcls, hstr, fW1, fb1, fW2, fb2, out);
}
